// PILNet_Conv_34986803593905
// MI455X (gfx1250) — matrix operand order repaired and checked by reading, not compiled, not run on hardware
//
#include <hip/hip_runtime.h>
#include <hip/hip_bf16.h>

// ---------------------------------------------------------------------------
// PILNet conv block, fused for MI455X (gfx1250, wave32, WMMA bf16 16x16x32).
// ---------------------------------------------------------------------------

#define N_NODES 20000
#define N_EDGES 320000
#define NF 128
#define EF 32
#define CF 3
#define HID 512

typedef __bf16 bf16_t;
typedef __attribute__((ext_vector_type(16))) __bf16 v16bf;
typedef __attribute__((ext_vector_type(8)))  __bf16 v8bf;
typedef __attribute__((ext_vector_type(8)))  float  v8f;
typedef __attribute__((ext_vector_type(4)))  float  v4f;

#if defined(__HIP_DEVICE_COMPILE__) && \
    __has_builtin(__builtin_amdgcn_global_load_async_to_lds_b128) && \
    __has_builtin(__builtin_amdgcn_s_wait_asynccnt)
#define HAVE_ASYNC_LDS 1
#else
#define HAVE_ASYNC_LDS 0
#endif

__device__ __forceinline__ float celu1(float x) {
    // jax.nn.celu with alpha=1: x>0 ? x : exp(x)-1
    return x > 0.0f ? x : (__expf(x) - 1.0f);
}

__device__ __forceinline__ v8f wmma_bf16(v16bf a, v16bf b, v8f c) {
    return __builtin_amdgcn_wmma_f32_16x16x32_bf16(
        /*neg_a=*/false, a, /*neg_b=*/false, b,
        /*c_mod=*/(short)0, c, /*reuse_a=*/false, /*reuse_b=*/false);
}

// A-fragment (16x32 bf16, M=row; lane<16 -> K {0..7,16..23}, lane>=16 -> {8..15,24..31})
// built from an fp32 row. k0 = ktile*32 + (lane>>4)*8.
__device__ __forceinline__ v16bf a_frag_f32(const float* __restrict__ row, int k0) {
    v4f x0 = *(const v4f*)(row + k0);
    v4f x1 = *(const v4f*)(row + k0 + 4);
    v4f x2 = *(const v4f*)(row + k0 + 16);
    v4f x3 = *(const v4f*)(row + k0 + 20);
    v16bf a;
#pragma unroll
    for (int i = 0; i < 4; ++i) {
        a[i]      = (bf16_t)x0[i];
        a[4 + i]  = (bf16_t)x1[i];
        a[8 + i]  = (bf16_t)x2[i];
        a[12 + i] = (bf16_t)x3[i];
    }
    return a;
}

// A-fragment from a bf16 row (e.g. LDS, padded stride), k0 as above.
__device__ __forceinline__ v16bf a_frag_bf16(const bf16_t* __restrict__ row, int k0) {
    v8bf lo = *(const v8bf*)(row + k0);
    v8bf hi = *(const v8bf*)(row + k0 + 16);
    v16bf a;
#pragma unroll
    for (int i = 0; i < 8; ++i) { a[i] = lo[i]; a[8 + i] = hi[i]; }
    return a;
}

// B-fragment (32x16 bf16): lane = column n; lane<16 holds K {0..7,16..23}, lane>=16 K {8..15,24..31}.
// WT stored transposed [N][K] bf16 so a fragment is two contiguous 8-bf16 runs (k0.., k0+16..) per lane.
__device__ __forceinline__ v16bf b_frag(const bf16_t* __restrict__ WT, int K,
                                        int ntile, int ktile, int lane) {
    int n  = ntile * 16 + (lane & 15);
    int k0 = ktile * 32 + (lane >> 4) * 8;
    const bf16_t* p = WT + (long)n * K + k0;
    v8bf lo = *(const v8bf*)p;
    v8bf hi = *(const v8bf*)(p + 16);
    v16bf b;
#pragma unroll
    for (int i = 0; i < 8; ++i) { b[i] = lo[i]; b[8 + i] = hi[i]; }
    return b;
}

// ---------------------------------------------------------------------------
// Prep kernels
// ---------------------------------------------------------------------------
__global__ void convert_wT_kernel(const float* __restrict__ W, bf16_t* __restrict__ WT,
                                  int K, int N) {
    int total = K * N;
    for (int idx = blockIdx.x * blockDim.x + threadIdx.x; idx < total;
         idx += gridDim.x * blockDim.x) {
        int n = idx / K, k = idx - n * K;
        WT[idx] = (bf16_t)W[(long)k * N + n];
    }
}

__global__ void zero_f32_kernel(float* __restrict__ p, int n) {
    for (int i = blockIdx.x * blockDim.x + threadIdx.x; i < n;
         i += gridDim.x * blockDim.x)
        p[i] = 0.0f;
}

// ---------------------------------------------------------------------------
// c = celu(cfeats @ Wc + bc)   (K=3, pure VALU)
// ---------------------------------------------------------------------------
__global__ void nodec_kernel(const float* __restrict__ cf, const float* __restrict__ Wc,
                             const float* __restrict__ bc, float* __restrict__ c) {
    const int total = N_NODES * HID;
    for (int idx = blockIdx.x * blockDim.x + threadIdx.x; idx < total;
         idx += gridDim.x * blockDim.x) {
        int row = idx >> 9;
        int col = idx & (HID - 1);
        float v = cf[row * 3 + 0] * Wc[col]
                + cf[row * 3 + 1] * Wc[HID + col]
                + cf[row * 3 + 2] * Wc[2 * HID + col]
                + bc[col];
        c[idx] = celu1(v);
    }
}

// ---------------------------------------------------------------------------
// h = celu(hfeats @ Wn + bn)   via WMMA, one wave per 16-node tile
// ---------------------------------------------------------------------------
__global__ void nodeh_kernel(const float* __restrict__ hfeats,
                             const bf16_t* __restrict__ WnT,
                             const float* __restrict__ bn,
                             float* __restrict__ h) {
    int wv   = threadIdx.x >> 5;
    int lane = threadIdx.x & 31;
    int tile = blockIdx.x * (blockDim.x >> 5) + wv;
    if (tile >= N_NODES / 16) return;
    int m = lane & 15, half = lane >> 4;
    int nbase = tile * 16;

    const float* arow = hfeats + (long)(nbase + m) * NF;
    v16bf afr[4];
#pragma unroll
    for (int t = 0; t < 4; ++t) afr[t] = a_frag_f32(arow, t * 32 + half * 8);

    for (int j = 0; j < HID / 16; ++j) {
        v8f acc = {};
#pragma unroll
        for (int t = 0; t < 4; ++t)
            acc = wmma_bf16(afr[t], b_frag(WnT, NF, j, t, lane), acc);
        int col = j * 16 + m;
        float bias = bn[col];
#pragma unroll
        for (int r = 0; r < 8; ++r) {
            int row = nbase + half * 8 + r;
            h[(long)row * HID + col] = celu1(acc[r] + bias);
        }
    }
}

// ---------------------------------------------------------------------------
// Edge pass (bandwidth-dominant):
//   stage A: e = celu(efeats@We+be)  (K=32 -> one WMMA per 16x16 tile) -> LDS bf16
//   stage B: one edge per iteration, whole wave: lane owns 16 contiguous columns
//            -> gathered h[src]/c[src]/c[dst] rows become contiguous 2KB bursts,
//               atomic segment-sum writes contiguous as well.
// One wave per 16-edge tile, 2 waves per block.
// ---------------------------------------------------------------------------
#define LDK  (HID + 8)   // bf16 elements per padded LDS row (16B-aligned stride)
#define EFLD 36          // fp32 elements per padded efeats LDS row

__global__ void edge_kernel(const float* __restrict__ efeats,
                            const int* __restrict__ src, const int* __restrict__ dst,
                            const bf16_t* __restrict__ WeT, const float* __restrict__ be,
                            const float* __restrict__ h, const float* __restrict__ c,
                            float* __restrict__ kbuf) {
    __shared__ __align__(16) bf16_t se[2][16 * LDK];
    __shared__ __align__(16) float  sef[2][16 * EFLD];
    int wv   = threadIdx.x >> 5;
    int lane = threadIdx.x & 31;
    int tile = blockIdx.x * 2 + wv;  // exact: 20000 tiles
    int m = lane & 15, half = lane >> 4;
    int ebase = tile * 16;
    bf16_t* S   = se[wv];
    float*  EFS = sef[wv];

    // --- stage efeats tile (16 x 32 fp32 = 2KB) into LDS; async path if available
    {
        const float* g = efeats + (long)(ebase + m) * EF + half * 16;
        float* l = EFS + m * EFLD + half * 16;
#if HAVE_ASYNC_LDS
        typedef int vi4 __attribute__((vector_size(16)));
        __attribute__((address_space(1))) vi4* gp =
            (__attribute__((address_space(1))) vi4*)g;
        __attribute__((address_space(3))) vi4* lp =
            (__attribute__((address_space(3))) vi4*)l;
        __builtin_amdgcn_global_load_async_to_lds_b128(gp, lp, 0, 0);
        __builtin_amdgcn_global_load_async_to_lds_b128(gp, lp, 16, 0);
        __builtin_amdgcn_global_load_async_to_lds_b128(gp, lp, 32, 0);
        __builtin_amdgcn_global_load_async_to_lds_b128(gp, lp, 48, 0);
        __builtin_amdgcn_s_wait_asynccnt(0);
#else
#pragma unroll
        for (int q = 0; q < 4; ++q)
            *(v4f*)(l + q * 4) = *(const v4f*)(g + q * 4);
#endif
    }

    // --- stage A: compute e tile via WMMA, celu, store bf16 into LDS
    v16bf a = a_frag_f32(EFS + m * EFLD, half * 8);
    for (int j = 0; j < HID / 16; ++j) {
        v8f acc = {};
        acc = wmma_bf16(a, b_frag(WeT, EF, j, 0, lane), acc);
        int col = j * 16 + m;
        float bias = be[col];
#pragma unroll
        for (int r = 0; r < 8; ++r)
            S[(half * 8 + r) * LDK + col] = (bf16_t)celu1(acc[r] + bias);
    }

    // --- stage B: per-edge wave-wide gather / message / scatter-add
    int eidx = ebase + (lane & 15);
    int s_l = src[eidx];
    int d_l = dst[eidx];
    int col0 = lane * 16;

    for (int i = 0; i < 16; ++i) {
        int s = __shfl(s_l, i, 32);
        int d = __shfl(d_l, i, 32);
        if (i + 1 < 16) {
            int sn = __shfl(s_l, i + 1, 32);
            int dn = __shfl(d_l, i + 1, 32);
            __builtin_prefetch(h + (long)sn * HID + col0, 0, 1);
            __builtin_prefetch(c + (long)sn * HID + col0, 0, 1);
            __builtin_prefetch(c + (long)dn * HID + col0, 0, 1);
        }
        const float* hs = h + (long)s * HID + col0;
        const float* cs = c + (long)s * HID + col0;
        const float* cd = c + (long)d * HID + col0;
        float*       kd = kbuf + (long)d * HID + col0;
        const bf16_t* er = S + i * LDK + col0;
        v8bf e0 = *(const v8bf*)er;
        v8bf e1 = *(const v8bf*)(er + 8);
#pragma unroll
        for (int q = 0; q < 4; ++q) {
            v4f hv  = *(const v4f*)(hs + q * 4);
            v4f csv = *(const v4f*)(cs + q * 4);
            v4f cdv = *(const v4f*)(cd + q * 4);
#pragma unroll
            for (int u = 0; u < 4; ++u) {
                int idx = q * 4 + u;
                float ev = (idx < 8) ? (float)e0[idx & 7] : (float)e1[idx & 7];
                float x = fabsf(cdv[u] - csv[u]) * (hv[u] * ev);
                atomicAdd(kd + idx, x);
            }
        }
    }
}

// ---------------------------------------------------------------------------
// h_out = hfeats + celu(celu(k@Wn1+bn1)@Wn2+bn2); two chained WMMA GEMMs with
// bf16 LDS staging of the celu'd intermediate. One wave per 16-node tile.
// ---------------------------------------------------------------------------
__global__ void hout_kernel(const float* __restrict__ kbuf,
                            const bf16_t* __restrict__ Wn1T, const float* __restrict__ bn1,
                            const bf16_t* __restrict__ Wn2T, const float* __restrict__ bn2,
                            const float* __restrict__ hfeats, float* __restrict__ hout) {
    __shared__ __align__(16) bf16_t sb[2][16 * LDK];
    int wv   = threadIdx.x >> 5;
    int lane = threadIdx.x & 31;
    int tile = blockIdx.x * 2 + wv;  // exact: 1250 tiles
    int m = lane & 15, half = lane >> 4;
    int nbase = tile * 16;
    bf16_t* S = sb[wv];

    // Stage 1: t1 = celu(k @ Wn1 + bn1) -> LDS (bf16)
    const float* arow = kbuf + (long)(nbase + m) * HID;
    v16bf afr[16];
#pragma unroll
    for (int t = 0; t < 16; ++t) afr[t] = a_frag_f32(arow, t * 32 + half * 8);

    for (int j = 0; j < HID / 16; ++j) {
        v8f acc = {};
#pragma unroll
        for (int t = 0; t < 16; ++t)
            acc = wmma_bf16(afr[t], b_frag(Wn1T, HID, j, t, lane), acc);
        int col = j * 16 + m;
        float bias = bn1[col];
#pragma unroll
        for (int r = 0; r < 8; ++r)
            S[(half * 8 + r) * LDK + col] = (bf16_t)celu1(acc[r] + bias);
    }
    __syncthreads();

    // Stage 2: h_out = hfeats + celu(t1 @ Wn2 + bn2)
#pragma unroll
    for (int t = 0; t < 16; ++t) afr[t] = a_frag_bf16(S + m * LDK, t * 32 + half * 8);

    for (int j = 0; j < NF / 16; ++j) {
        v8f acc = {};
#pragma unroll
        for (int t = 0; t < 16; ++t)
            acc = wmma_bf16(afr[t], b_frag(Wn2T, HID, j, t, lane), acc);
        int col = j * 16 + m;
        float bias = bn2[col];
#pragma unroll
        for (int r = 0; r < 8; ++r) {
            int row = nbase + half * 8 + r;
            long o = (long)row * NF + col;
            hout[o] = hfeats[o] + celu1(acc[r] + bias);
        }
    }
}

// ---------------------------------------------------------------------------
// e_out = efeats + celu(e @ Weo + beo), recomputing e = celu(efeats@We+be).
// One wave per 16-edge tile; celu'd e staged as bf16 in LDS.
// ---------------------------------------------------------------------------
__global__ void eout_kernel(const float* __restrict__ efeats,
                            const bf16_t* __restrict__ WeT, const float* __restrict__ be,
                            const bf16_t* __restrict__ WeoT, const float* __restrict__ beo,
                            float* __restrict__ eout) {
    __shared__ __align__(16) bf16_t sb[2][16 * LDK];
    int wv   = threadIdx.x >> 5;
    int lane = threadIdx.x & 31;
    int tile = blockIdx.x * 2 + wv;  // exact: 20000 tiles
    int m = lane & 15, half = lane >> 4;
    int ebase = tile * 16;
    bf16_t* S = sb[wv];

    v16bf a = a_frag_f32(efeats + (long)(ebase + m) * EF, half * 8);

    for (int j = 0; j < HID / 16; ++j) {
        v8f acc = {};
        acc = wmma_bf16(a, b_frag(WeT, EF, j, 0, lane), acc);
        int col = j * 16 + m;
        float bias = be[col];
#pragma unroll
        for (int r = 0; r < 8; ++r)
            S[(half * 8 + r) * LDK + col] = (bf16_t)celu1(acc[r] + bias);
    }
    __syncthreads();

    v16bf afr[16];
#pragma unroll
    for (int t = 0; t < 16; ++t) afr[t] = a_frag_bf16(S + m * LDK, t * 32 + half * 8);

    for (int j = 0; j < EF / 16; ++j) {
        v8f acc = {};
#pragma unroll
        for (int t = 0; t < 16; ++t)
            acc = wmma_bf16(afr[t], b_frag(WeoT, HID, j, t, lane), acc);
        int col = j * 16 + m;
        float bias = beo[col];
#pragma unroll
        for (int r = 0; r < 8; ++r) {
            int row = ebase + half * 8 + r;
            long o = (long)row * EF + col;
            eout[o] = efeats[o] + celu1(acc[r] + bias);
        }
    }
}

// ---------------------------------------------------------------------------
// c_out = cfeats + celu(c @ Wco + bco)  (N=3, tiny: one thread per node)
// ---------------------------------------------------------------------------
__global__ void cout_kernel(const float* __restrict__ cbuf, const float* __restrict__ Wco,
                            const float* __restrict__ bco, const float* __restrict__ cfeats,
                            float* __restrict__ cout) {
    int i = blockIdx.x * blockDim.x + threadIdx.x;
    if (i >= N_NODES) return;
    const float* crow = cbuf + (long)i * HID;
    float s0 = 0.f, s1 = 0.f, s2 = 0.f;
    for (int t = 0; t < HID; ++t) {
        float v = crow[t];
        s0 += v * Wco[t * 3 + 0];
        s1 += v * Wco[t * 3 + 1];
        s2 += v * Wco[t * 3 + 2];
    }
    cout[i * 3 + 0] = cfeats[i * 3 + 0] + celu1(s0 + bco[0]);
    cout[i * 3 + 1] = cfeats[i * 3 + 1] + celu1(s1 + bco[1]);
    cout[i * 3 + 2] = cfeats[i * 3 + 2] + celu1(s2 + bco[2]);
}

// ---------------------------------------------------------------------------
extern "C" void kernel_launch(void* const* d_in, const int* in_sizes, int n_in,
                              void* d_out, int out_size, void* d_ws, size_t ws_size,
                              hipStream_t stream) {
    const float* hfeats = (const float*)d_in[0];
    const float* cfeats = (const float*)d_in[1];
    const float* efeats = (const float*)d_in[2];
    const int*   src    = (const int*)d_in[3];
    const int*   dst    = (const int*)d_in[4];
    const float* Wn  = (const float*)d_in[5];  const float* bn  = (const float*)d_in[6];
    const float* We  = (const float*)d_in[7];  const float* be  = (const float*)d_in[8];
    const float* Wc  = (const float*)d_in[9];  const float* bc  = (const float*)d_in[10];
    const float* Wn1 = (const float*)d_in[11]; const float* bn1 = (const float*)d_in[12];
    const float* Wn2 = (const float*)d_in[13]; const float* bn2 = (const float*)d_in[14];
    const float* Weo = (const float*)d_in[15]; const float* beo = (const float*)d_in[16];
    const float* Wco = (const float*)d_in[17]; const float* bco = (const float*)d_in[18];

    // Workspace carve-out (256B aligned slices)
    char* wsb = (char*)d_ws;
    size_t off = 0;
    auto carve = [&](size_t bytes) {
        void* p = wsb + off;
        off = (off + bytes + 255) & ~(size_t)255;
        return p;
    };
    bf16_t* WnT  = (bf16_t*)carve((size_t)HID * NF  * 2);  // [512][128]
    bf16_t* WeT  = (bf16_t*)carve((size_t)HID * EF  * 2);  // [512][32]
    bf16_t* Wn1T = (bf16_t*)carve((size_t)HID * HID * 2);  // [512][512]
    bf16_t* Wn2T = (bf16_t*)carve((size_t)NF  * HID * 2);  // [128][512]
    bf16_t* WeoT = (bf16_t*)carve((size_t)EF  * HID * 2);  // [32][512]
    float*  hbuf = (float*)carve((size_t)N_NODES * HID * 4);
    float*  cbuf = (float*)carve((size_t)N_NODES * HID * 4);
    float*  kbuf = (float*)carve((size_t)N_NODES * HID * 4);

    float* hout = (float*)d_out;
    float* cout = hout + (size_t)N_NODES * NF;
    float* eout = cout + (size_t)N_NODES * CF;

    // Weight conversion (transposed bf16) + zero segment-sum accumulator
    convert_wT_kernel<<<128, 256, 0, stream>>>(Wn,  WnT,  NF,  HID);
    convert_wT_kernel<<<32,  256, 0, stream>>>(We,  WeT,  EF,  HID);
    convert_wT_kernel<<<512, 256, 0, stream>>>(Wn1, Wn1T, HID, HID);
    convert_wT_kernel<<<128, 256, 0, stream>>>(Wn2, Wn2T, HID, NF);
    convert_wT_kernel<<<32,  256, 0, stream>>>(Weo, WeoT, HID, EF);
    zero_f32_kernel<<<2048, 256, 0, stream>>>(kbuf, N_NODES * HID);

    // Feature expansions
    nodeh_kernel<<<(N_NODES / 16 + 3) / 4, 128, 0, stream>>>(hfeats, WnT, bn, hbuf);
    nodec_kernel<<<2048, 256, 0, stream>>>(cfeats, Wc, bc, cbuf);

    // Edge messages + segment sum (bandwidth-dominant)
    edge_kernel<<<N_EDGES / 16 / 2, 64, 0, stream>>>(efeats, src, dst, WeT, be,
                                                     hbuf, cbuf, kbuf);

    // Output reductions (fused residual + celu epilogues)
    hout_kernel<<<N_NODES / 16 / 2, 64, 0, stream>>>(kbuf, Wn1T, bn1, Wn2T, bn2,
                                                     hfeats, hout);
    eout_kernel<<<N_EDGES / 16 / 2, 64, 0, stream>>>(efeats, WeT, be, WeoT, beo, eout);
    cout_kernel<<<(N_NODES + 255) / 256, 256, 0, stream>>>(cbuf, Wco, bco, cfeats, cout);
}
